// GRU_70901320122737
// MI455X (gfx1250) — matrix operand order repaired and checked by reading, not compiled, not run on hardware
//
#include <hip/hip_runtime.h>

// ---------------------------------------------------------------------------
// GRU LM for MI455X (gfx1250):
//   - bf16 WMMA GEMMs with TDM (tensor_load_to_lds) double-buffered LDS staging
//   - persistent grid-synced GRU scan (WMMA recurrence)
// ---------------------------------------------------------------------------

typedef __attribute__((ext_vector_type(16))) __bf16 bf16x16;
typedef __attribute__((ext_vector_type(8)))  __bf16 bf16x8;
typedef __attribute__((ext_vector_type(8)))  float  f32x8;
typedef __attribute__((ext_vector_type(4)))  unsigned int u32x4;
typedef __attribute__((ext_vector_type(8)))  int          i32x8;
typedef __attribute__((ext_vector_type(4)))  int          i32x4;

constexpr int S = 128, B = 32, H = 512, V = 10000, L = 2;
constexpr int SB  = S * B;          // 4096 rows
constexpr int NWG = 16;             // workgroups in persistent scan kernel
constexpr int TPB = 256;            // 8 waves per workgroup (wave32)
constexpr int KC  = 64;             // GEMM K chunk staged per TDM transfer
constexpr int LDT = 72;             // padded LDS row pitch in elements (144 B)

// ---------------- WMMA fragment helpers (wave32, 16x16x32 bf16) -------------
// A: row-major [M,K] bf16. Lane (0..15): M=lane, K in {k0..k0+7, k0+16..k0+23};
// lane (16..31): same M range, K offset +8.  Two 16B contiguous loads/lane.
__device__ __forceinline__ bf16x16 load_a_frag(const __bf16* A, int lda,
                                               int m0, int k0) {
  const int lane = threadIdx.x & 31;
  const int m  = m0 + (lane & 15);
  const int kb = k0 + ((lane & 16) ? 8 : 0);
  const __bf16* p = A + (size_t)m * lda + kb;
  bf16x8 lo = *(const bf16x8*)(p);
  bf16x8 hi = *(const bf16x8*)(p + 16);
  bf16x16 r;
#pragma unroll
  for (int i = 0; i < 8; ++i) { r[i] = lo[i]; r[8 + i] = hi[i]; }
  return r;
}

// B stored TRANSPOSED: Bt row-major [N,K] bf16. Lane (0..15): N=lane,
// K in {k0..k0+7, k0+16..k0+23}; lane (16..31): N=lane-16, K offset +8.
__device__ __forceinline__ bf16x16 load_b_frag(const __bf16* Bt, int ldb,
                                               int n0, int k0) {
  const int lane = threadIdx.x & 31;
  const int n  = n0 + (lane & 15);
  const int kb = k0 + ((lane & 16) ? 8 : 0);
  const __bf16* p = Bt + (size_t)n * ldb + kb;
  bf16x8 lo = *(const bf16x8*)(p);
  bf16x8 hi = *(const bf16x8*)(p + 16);
  bf16x16 r;
#pragma unroll
  for (int i = 0; i < 8; ++i) { r[i] = lo[i]; r[8 + i] = hi[i]; }
  return r;
}

__device__ __forceinline__ f32x8 wmma_bf16(bf16x16 a, bf16x16 b, f32x8 c) {
  return __builtin_amdgcn_wmma_f32_16x16x32_bf16(false, a, false, b,
                                                 (short)0, c, false, false);
}

// ---------------- TDM: 2D tile (rows x KC bf16) global -> LDS ---------------
// D# per CDNA5 ISA. data_size=2B. LDS padding: after every
// 32 DWORDs (one 128B row) insert 4 DWORDs -> 144B row pitch (LDT elements),
// staggering rows across LDS banks for conflict-free fragment reads.
__device__ __forceinline__ void tdm_load_2d(unsigned lds_off,
                                            unsigned long long gaddr,
                                            unsigned tensor_d0, unsigned tensor_d1,
                                            unsigned tile_d1, unsigned stride_d0) {
  u32x4 g0;
  g0[0] = 1u;                                            // count=1 (user mode)
  g0[1] = lds_off;                                       // lds_addr (bytes)
  g0[2] = (unsigned)(gaddr & 0xFFFFFFFFull);             // global_addr[31:0]
  g0[3] = (unsigned)((gaddr >> 32) & 0x1FFFFFFull)       // global_addr[56:32]
          | (2u << 30);                                  // type=2 ("image")
  i32x8 g1;
  g1[0] = (int)((1u << 16)      // data_size = 2 bytes
              | (1u << 20)      // pad_enable
              | (4u << 22)      // pad_interval: 32 DWORDs
              | (3u << 25));    // pad_amount:   4 DWORDs
  g1[1] = (int)((tensor_d0 & 0xFFFFu) << 16);            // tensor_dim0[15:0]
  g1[2] = (int)((tensor_d0 >> 16) | ((tensor_d1 & 0xFFFFu) << 16));
  g1[3] = (int)((tensor_d1 >> 16) | ((unsigned)KC << 16)); // tile_dim0 = KC
  g1[4] = (int)(tile_d1 & 0xFFFFu);                      // tile_dim1 (tile_dim2=0)
  g1[5] = (int)stride_d0;                                // tensor_dim0_stride lo
  g1[6] = 0;                                             // stride hi / dim1_stride
  g1[7] = 0;
  i32x4 g2 = {0, 0, 0, 0};
  i32x4 g3 = {0, 0, 0, 0};
  i32x8 g4 = {0, 0, 0, 0, 0, 0, 0, 0};
  __builtin_amdgcn_tensor_load_to_lds(g0, g1, g2, g3, g4, 0);
}

// ---------------- GEMM: C[M,N] = A[M,K] * Bt[N,K]^T + bias ------------------
// Block tile 32x256 (8 waves, each 16x64 = 4 accumulators). K staged in KC=64
// chunks via TDM, double-buffered: chunk i+1's DMA overlaps chunk i's WMMAs.
__global__ __launch_bounds__(TPB)
void gemm_bf16_tdm(const __bf16* __restrict__ A, int lda,
                   const __bf16* __restrict__ Bt, int ldb,
                   const float* __restrict__ bias,
                   float* __restrict__ C, int ldc,
                   int M, int N, int K) {
  __shared__ __bf16 sA[2][32 * LDT];    //  9.2 KB
  __shared__ __bf16 sB[2][256 * LDT];   // 73.7 KB
  const int wave = threadIdx.x >> 5;
  const int lane = threadIdx.x & 31;
  const int wm = wave >> 2, wn = wave & 3;
  const int m0 = blockIdx.y * 32;
  const int n0 = blockIdx.x * 256;
  const int nchunks = K / KC;

  auto issue_chunk = [&](int ci, int bufi) {
    const int k0 = ci * KC;
    tdm_load_2d((unsigned)(unsigned long long)&sA[bufi][0],
                (unsigned long long)(const void*)(A + (size_t)m0 * lda + k0),
                (unsigned)(K - k0), (unsigned)(M - m0), 32u, (unsigned)lda);
    tdm_load_2d((unsigned)(unsigned long long)&sB[bufi][0],
                (unsigned long long)(const void*)(Bt + (size_t)n0 * ldb + k0),
                (unsigned)(K - k0), (unsigned)(N - n0), 256u, (unsigned)ldb);
  };

  if (wave == 0) issue_chunk(0, 0);

  f32x8 acc[4] = {};
  for (int ci = 0; ci < nchunks; ++ci) {
    const int bufi = ci & 1;
    if (wave == 0) {
      if (ci + 1 < nchunks) {
        issue_chunk(ci + 1, bufi ^ 1);
        __builtin_amdgcn_s_wait_tensorcnt(2);  // chunk ci's two DMAs done
      } else {
        __builtin_amdgcn_s_wait_tensorcnt(0);
      }
    }
    __syncthreads();
#pragma unroll
    for (int ks = 0; ks < KC; ks += 32) {
      bf16x16 a = load_a_frag(&sA[bufi][0], LDT, wm * 16, ks);
#pragma unroll
      for (int j = 0; j < 4; ++j) {
        bf16x16 b = load_b_frag(&sB[bufi][0], LDT, wn * 64 + j * 16, ks);
        acc[j] = wmma_bf16(a, b, acc[j]);
      }
    }
    __syncthreads();
  }

  const int nlo = lane & 15;
  const int mB = m0 + wm * 16 + ((lane & 16) ? 8 : 0);
#pragma unroll
  for (int j = 0; j < 4; ++j) {
    const int col = n0 + wn * 64 + j * 16 + nlo;
    if (col < N) {
      const float bv = bias ? bias[col] : 0.f;
#pragma unroll
      for (int v = 0; v < 8; ++v)
        C[(size_t)(mB + v) * ldc + col] = acc[j][v] + bv;
    }
  }
}

// ---------------- Software grid barrier (agent scope) -----------------------
__device__ __forceinline__ void grid_sync(unsigned* bar) {
  __threadfence();
  __syncthreads();
  if (threadIdx.x == 0) {
    unsigned gen = __hip_atomic_load(bar + 1, __ATOMIC_RELAXED, __HIP_MEMORY_SCOPE_AGENT);
    unsigned a = __hip_atomic_fetch_add(bar, 1u, __ATOMIC_ACQ_REL, __HIP_MEMORY_SCOPE_AGENT);
    if (a == (unsigned)(NWG - 1)) {
      __hip_atomic_store(bar, 0u, __ATOMIC_RELAXED, __HIP_MEMORY_SCOPE_AGENT);
      __hip_atomic_fetch_add(bar + 1, 1u, __ATOMIC_RELEASE, __HIP_MEMORY_SCOPE_AGENT);
    } else {
      while (__hip_atomic_load(bar + 1, __ATOMIC_ACQUIRE, __HIP_MEMORY_SCOPE_AGENT) == gen)
        __builtin_amdgcn_s_sleep(2);
    }
  }
  __syncthreads();
}

// ---------------- Persistent GRU scan (one layer, all timesteps) ------------
// ax:   f32 [3][S*B][H]  precomputed x@Wx + bx (gate planes)
// Ut:   bf16 [3][H][H]   recurrent weights, transposed to [out,in]
// h:    f32 [B][H]       carried state (in/out)
__global__ __launch_bounds__(TPB)
void gru_scan(const float* __restrict__ ax, const __bf16* __restrict__ Ut,
              float* __restrict__ h, __bf16* __restrict__ h_bf,
              float* __restrict__ z_buf, __bf16* __restrict__ rh_bf,
              __bf16* __restrict__ xout, unsigned* __restrict__ bar) {
  const int wv   = blockIdx.x * (TPB / 32) + (threadIdx.x >> 5);  // 0..127
  const int lane = threadIdx.x & 31;
  const int tid  = blockIdx.x * TPB + threadIdx.x;                // 0..4095
  const size_t SBH = (size_t)SB * H;

  for (int t = 0; t < S; ++t) {
    // ---- phase A: h (f32) -> h_bf (bf16) for WMMA feed ----
    for (int i = tid; i < B * H; i += NWG * TPB) h_bf[i] = (__bf16)h[i];
    grid_sync(bar);

    // ---- phase B: r,z = sigmoid(ax_g + h@U_g); stash z and r*h ----
    {
      const int mt = wv & 1, nt = wv >> 1;   // 2 x 64 tiles over [32,1024]
      const int g  = nt >> 5;                // gate 0 (r) or 1 (z)
      const int n0 = (nt & 31) * 16, m0 = mt * 16;
      f32x8 acc = {};
      for (int k0 = 0; k0 < H; k0 += 32) {
        bf16x16 a = load_a_frag(h_bf, H, m0, k0);
        bf16x16 b = load_b_frag(Ut + (size_t)g * H * H, H, n0, k0);
        acc = wmma_bf16(a, b, acc);
      }
      const int n  = n0 + (lane & 15);
      const int mB = m0 + ((lane & 16) ? 8 : 0);
      const float* axg = ax + (size_t)g * SBH + (size_t)t * B * H;
#pragma unroll
      for (int v = 0; v < 8; ++v) {
        const int row = mB + v;
        const float pre = acc[v] + axg[(size_t)row * H + n];
        const float sg  = 1.f / (1.f + __expf(-pre));
        if (g == 0) rh_bf[row * H + n] = (__bf16)(sg * h[row * H + n]);
        else        z_buf[row * H + n] = sg;
      }
    }
    grid_sync(bar);

    // ---- phase C: h_hat = tanh(ax_2 + (r*h)@U_2); h = (1-z)h + z*h_hat ----
    if (wv < 64) {
      const int mt = wv & 1, nt = wv >> 1;   // 2 x 32 tiles over [32,512]
      const int n0 = nt * 16, m0 = mt * 16;
      f32x8 acc = {};
      for (int k0 = 0; k0 < H; k0 += 32) {
        bf16x16 a = load_a_frag(rh_bf, H, m0, k0);
        bf16x16 b = load_b_frag(Ut + 2 * (size_t)H * H, H, n0, k0);
        acc = wmma_bf16(a, b, acc);
      }
      const int n  = n0 + (lane & 15);
      const int mB = m0 + ((lane & 16) ? 8 : 0);
      const float* ax2 = ax + 2 * SBH + (size_t)t * B * H;
#pragma unroll
      for (int v = 0; v < 8; ++v) {
        const int row = mB + v;
        const float hc = tanhf(acc[v] + ax2[(size_t)row * H + n]);
        const float z  = z_buf[row * H + n];
        const float hn = (1.f - z) * h[row * H + n] + z * hc;
        h[row * H + n] = hn;
        xout[((size_t)(t * B + row)) * H + n] = (__bf16)hn;
      }
    }
    grid_sync(bar);
  }
}

// ---------------- Prep kernels ----------------------------------------------
__global__ __launch_bounds__(TPB)
void k_transpose_bf16(const float* __restrict__ in, __bf16* __restrict__ outp, int nmat) {
  size_t idx = (size_t)blockIdx.x * blockDim.x + threadIdx.x;
  if (idx >= (size_t)nmat * H * H) return;
  const int m = (int)(idx / (H * H));
  const int r = (int)(idx % (H * H));
  const int o = r / H, i = r % H;
  outp[idx] = (__bf16)in[(size_t)m * H * H + (size_t)i * H + o];  // [o][i] <- [i][o]
}

__global__ __launch_bounds__(TPB)
void k_cvt_bf16(const float* __restrict__ in, __bf16* __restrict__ outp, size_t n) {
  size_t idx = (size_t)blockIdx.x * blockDim.x + threadIdx.x;
  if (idx < n) outp[idx] = (__bf16)in[idx];
}

__global__ __launch_bounds__(TPB)
void k_embed(const int* __restrict__ tokens, const float* __restrict__ emb,
             __bf16* __restrict__ x) {
  size_t idx = (size_t)blockIdx.x * blockDim.x + threadIdx.x;
  if (idx >= (size_t)SB * H) return;
  const int sb = (int)(idx / H), c = (int)(idx % H);
  const int tok = tokens[sb];
  x[idx] = (__bf16)emb[(size_t)tok * H + c];
}

// ---------------- Host orchestration ----------------------------------------
extern "C" void kernel_launch(void* const* d_in, const int* in_sizes, int n_in,
                              void* d_out, int out_size, void* d_ws, size_t ws_size,
                              hipStream_t stream) {
  const int*   tokens = (const int*)d_in[0];
  const float* h0     = (const float*)d_in[1];
  const float* emb    = (const float*)d_in[2];
  const float* Wx     = (const float*)d_in[3];
  const float* bx     = (const float*)d_in[4];
  const float* U      = (const float*)d_in[5];
  const float* Wy     = (const float*)d_in[6];
  const float* by     = (const float*)d_in[7];
  float* out = (float*)d_out;

  // workspace carve-up (~50 MB)
  char* p = (char*)d_ws;
  auto carve = [&](size_t bytes) { char* r = p; p += (bytes + 255) & ~(size_t)255; return (void*)r; };
  __bf16* xA   = (__bf16*)carve((size_t)SB * H * 2);
  __bf16* xB   = (__bf16*)carve((size_t)SB * H * 2);
  float*  axb  = (float*)carve((size_t)3 * SB * H * 4);
  __bf16* WxT  = (__bf16*)carve((size_t)L * 3 * H * H * 2);
  __bf16* UT   = (__bf16*)carve((size_t)L * 3 * H * H * 2);
  __bf16* WyB  = (__bf16*)carve((size_t)V * H * 2);
  float*  hbuf = (float*)carve((size_t)B * H * 4);
  __bf16* hbf  = (__bf16*)carve((size_t)B * H * 2);
  float*  zbuf = (float*)carve((size_t)B * H * 4);
  __bf16* rhbf = (__bf16*)carve((size_t)B * H * 2);
  unsigned* bar = (unsigned*)carve(256);

  // weight conversion (transpose Wx/U to [out,in]; Wy already B-transposed)
  {
    const size_t nW = (size_t)L * 3 * H * H;
    k_transpose_bf16<<<(int)((nW + TPB - 1) / TPB), TPB, 0, stream>>>(Wx, WxT, L * 3);
    k_transpose_bf16<<<(int)((nW + TPB - 1) / TPB), TPB, 0, stream>>>(U, UT, L * 3);
    const size_t nWy = (size_t)V * H;
    k_cvt_bf16<<<(int)((nWy + TPB - 1) / TPB), TPB, 0, stream>>>(Wy, WyB, nWy);
  }

  // embedding gather -> bf16 activations [S*B, H]
  k_embed<<<(SB * H) / TPB, TPB, 0, stream>>>(tokens, emb, xA);

  __bf16* x_in = xA;
  __bf16* x_out = xB;
  for (int l = 0; l < L; ++l) {
    // hoisted input projections: ax_g = x @ Wx[l,g] + bx[l,g]  (3x big GEMM)
    for (int g = 0; g < 3; ++g) {
      gemm_bf16_tdm<<<dim3((H + 255) / 256, SB / 32), TPB, 0, stream>>>(
          x_in, H, WxT + (size_t)(l * 3 + g) * H * H, H,
          bx + (size_t)(l * 3 + g) * H,
          axb + (size_t)g * SB * H, H, SB, H, H);
    }
    // init carried state, reset barrier, run persistent scan
    (void)hipMemcpyAsync(hbuf, h0 + (size_t)l * B * H, (size_t)B * H * 4,
                         hipMemcpyDeviceToDevice, stream);
    (void)hipMemsetAsync(bar, 0, 2 * sizeof(unsigned), stream);
    gru_scan<<<NWG, TPB, 0, stream>>>(axb, UT + (size_t)l * 3 * H * H,
                                      hbuf, hbf, zbuf, rhbf, x_out, bar);
    // h_final for this layer -> output tail
    (void)hipMemcpyAsync(out + (size_t)SB * V + (size_t)l * B * H, hbuf,
                         (size_t)B * H * 4, hipMemcpyDeviceToDevice, stream);
    __bf16* t = x_in; x_in = x_out; x_out = t;
  }

  // logits = tops @ Wy^T + by  -> d_out [4096, 10000]
  gemm_bf16_tdm<<<dim3((V + 255) / 256, SB / 32), TPB, 0, stream>>>(
      x_in, H, WyB, H, by, out, V, SB, V, H);
}
